// HMMLoss_75333726372055
// MI455X (gfx1250) — matrix operand order repaired and checked by reading, not compiled, not run on hardware
//
#include <hip/hip_runtime.h>

typedef __attribute__((ext_vector_type(16))) __bf16 v16bf;
typedef __attribute__((ext_vector_type(8)))  float  v8f;

#define H_      512
#define V_      32000
#define NS_     128
#define ROWS_   4096          // PH*TL = 128*32
#define TLEN_   32
#define NSPLIT_ 10
#define COLSPLIT_ 3200        // V_/NSPLIT_
#define PARTS_  (NSPLIT_*4)   // 40 partials per row (4 column-waves x 10 splits)
#define NCT_    (V_/16)       // 2000 column tiles
#define KSTEPS_ (H_/32)       // 16 wmma K-steps
#define MB_     32            // rows per block (2 x 16-row wave groups)

__device__ __forceinline__ __bf16 f2bf(float f) {
  unsigned u = __builtin_bit_cast(unsigned, f);
  unsigned r = u + 0x7FFFu + ((u >> 16) & 1u);      // round-to-nearest-even
  unsigned short h = (unsigned short)(r >> 16);
  return __builtin_bit_cast(__bf16, h);
}

// ---------------------------------------------------------------------------
// Pack W [H,V] f32 (row-major) into per-lane bf16 B-fragments:
// frag[(ct*16+kk)*32 + lane][j] = bf16( W[kk*32 + (lane&16) + j][ct*16 + (lane&15)] )
// so a lane's 16 bf16 (32B) are contiguous -> 2x global_load_b128 per K-step.
// ---------------------------------------------------------------------------
__global__ void hmm_pack_w(const float* __restrict__ W, __bf16* __restrict__ Wfrag) {
  int t = blockIdx.x * blockDim.x + threadIdx.x;    // < NCT_*16*32 = 1,024,000
  if (t >= NCT_ * KSTEPS_ * 32) return;
  int lane = t & 31;
  int kk   = (t >> 5) & (KSTEPS_ - 1);
  int ct   = t >> 9;
  int v    = ct * 16 + (lane & 15);
  int hb   = kk * 32 + (lane & 16);
  __bf16* dst = Wfrag + (size_t)t * 16;
#pragma unroll
  for (int j = 0; j < 16; ++j)
    dst[j] = f2bf(W[(size_t)(hb + j) * V_ + v]);
}

// ---------------------------------------------------------------------------
// Streaming GEMM + per-lane online softmax.
// Block = 256 threads (8 waves), owns 32 rows. Waves 0-3: rows 0-15,
// waves 4-7: rows 16-31 over the SAME columns -> B loads dedup in WGP$.
// grid = (ROWS/32, NSPLIT).
// ---------------------------------------------------------------------------
__global__ void __launch_bounds__(256)
hmm_gemm_softmax(const float* __restrict__ X, const float* __restrict__ bias,
                 const __bf16* __restrict__ Wfrag,
                 float* __restrict__ partM, float* __restrict__ partS) {
  __shared__ float xs[MB_ * H_];                    // 64 KB staged A rows
  const int rowbase = blockIdx.x * MB_;
  for (int i = threadIdx.x; i < MB_ * H_; i += 256)
    xs[i] = X[(size_t)rowbase * H_ + i];
  __syncthreads();

  const int wid   = threadIdx.x >> 5;
  const int lane  = threadIdx.x & 31;
  const int cwid  = wid & 3;                        // column-slot of this wave
  const int mhalf = wid >> 2;                       // 0: rows 0-15, 1: rows 16-31
  const int half8 = ((lane >> 4) & 1) * 8;

  // A fragments, register resident for all 50 tiles (contiguous k = 16*(lane/16) + j, as in Wfrag)
  v16bf afrag[KSTEPS_];
#pragma unroll
  for (int kk = 0; kk < KSTEPS_; ++kk) {
#pragma unroll
    for (int j = 0; j < 16; ++j) {
      int r  = j >> 1;
      int k0 = j + 2 * half8;                       // {0..15} (+16 hi half)
      int k  = kk * 32 + k0 + 0 * r;
      afrag[kk][j] = f2bf(xs[(mhalf * 16 + (lane & 15)) * H_ + k]);
    }
  }

  // Per-lane online softmax state (commutative across the columns a lane sees)
  float m8[8], s8[8];
#pragma unroll
  for (int i = 0; i < 8; ++i) { m8[i] = -__builtin_inff(); s8[i] = 0.0f; }

  const int colbase0 = blockIdx.y * COLSPLIT_ + cwid * 16;
  for (int tile = 0; tile < COLSPLIT_ / 64; ++tile) {      // 50 iterations
    const int col = colbase0 + tile * 64;
    const int ct  = col >> 4;
    const v16bf* Bp = reinterpret_cast<const v16bf*>(Wfrag)
                      + (size_t)ct * KSTEPS_ * 32 + lane;
    __builtin_prefetch((const void*)(Bp + (size_t)4 * KSTEPS_ * 32), 0, 1);

    v8f acc = {0.f, 0.f, 0.f, 0.f, 0.f, 0.f, 0.f, 0.f};
#pragma unroll
    for (int kk = 0; kk < KSTEPS_; ++kk) {
      v16bf bfrag = Bp[(size_t)kk * 32];
      acc = __builtin_amdgcn_wmma_f32_16x16x32_bf16(
          false, afrag[kk], false, bfrag, (short)0, acc, false, false);
    }

    const float bv = bias[col + (lane & 15)];
#pragma unroll
    for (int i = 0; i < 8; ++i) {                   // no cross-lane ops here
      float v  = acc[i] + bv;
      float nm = fmaxf(m8[i], v);
      s8[i] = s8[i] * __expf(m8[i] - nm) + __expf(v - nm);
      m8[i] = nm;
    }
  }

  // One-time cross-lane (m,s) merge within each 16-lane half
#pragma unroll
  for (int i = 0; i < 8; ++i) {
    float m = m8[i], s = s8[i];
#pragma unroll
    for (int off = 1; off < 16; off <<= 1) {
      float mo = __shfl_xor(m, off, 32);
      float so = __shfl_xor(s, off, 32);
      float nm = fmaxf(m, mo);
      s = s * __expf(m - nm) + so * __expf(mo - nm);
      m = nm;
    }
    m8[i] = m; s8[i] = s;
  }

  if ((lane & 15) == 0) {                           // lanes 0 (rows +0..7) & 16 (+8..15)
    int rb   = rowbase + mhalf * 16 + half8;
    int pidx = blockIdx.y * 4 + cwid;
#pragma unroll
    for (int i = 0; i < 8; ++i) {
      partM[(size_t)(rb + i) * PARTS_ + pidx] = m8[i];
      partS[(size_t)(rb + i) * PARTS_ + pidx] = s8[i];
    }
  }
}

// ---------------------------------------------------------------------------
// Merge partials -> logZ per row; fp32 dot for target logit; psk = t - logZ.
// ---------------------------------------------------------------------------
__global__ void hmm_merge_target(const float* __restrict__ X, const float* __restrict__ W,
                                 const float* __restrict__ bias, const int* __restrict__ target,
                                 const float* __restrict__ partM, const float* __restrict__ partS,
                                 float* __restrict__ psk) {
  int row = blockIdx.x * blockDim.x + threadIdx.x;
  if (row >= ROWS_) return;
  float m = -__builtin_inff();
  for (int j = 0; j < PARTS_; ++j) m = fmaxf(m, partM[(size_t)row * PARTS_ + j]);
  float s = 0.0f;
  for (int j = 0; j < PARTS_; ++j)
    s += partS[(size_t)row * PARTS_ + j] * __expf(partM[(size_t)row * PARTS_ + j] - m);
  float logZ = m + __logf(s);

  int tg = target[row];
  float dot = bias[tg];
  const float* xr = X + (size_t)row * H_;
#pragma unroll 8
  for (int h = 0; h < H_; ++h)
    dot = fmaf(xr[h], W[(size_t)h * V_ + tg], dot);
  psk[row] = dot - logZ;
}

// ---------------------------------------------------------------------------
// fwd_obs[p] = sum of psk over inclusive span [tgt_idx[p,0], tgt_idx[p,1]].
// ---------------------------------------------------------------------------
__global__ void hmm_span_sum(const float* __restrict__ psk, const int* __restrict__ tgt_idx,
                             float* __restrict__ fwd) {
  int p = threadIdx.x;
  if (p >= 128) return;
  int b = tgt_idx[p * 2], e = tgt_idx[p * 2 + 1];
  float s = 0.0f;
  for (int l = 0; l < TLEN_; ++l)
    if (l >= b && l <= e) s += psk[p * TLEN_ + l];
  fwd[p] = s;
}

// ---------------------------------------------------------------------------
// Tiny HMM backward scan (B=4, T=8, K=4, LS=4). One block of 128 threads.
// ---------------------------------------------------------------------------
__global__ void hmm_final(const float* __restrict__ fwd, const int* __restrict__ states,
                          const float* __restrict__ initlp, const float* __restrict__ trans,
                          const float* __restrict__ ext, const int* __restrict__ sidp,
                          float* __restrict__ out) {
  __shared__ int   stall[128][4];
  __shared__ int   sfA[128], slA[128];
  __shared__ float zArr[128], obsArr[128];
  __shared__ float beta[16], nbeta[16], sc0[16];
  const float NEGINF = -__builtin_inff();

  int r = threadIdx.x;                              // exactly 128 threads
  int s0 = states[r * 4 + 0], s1 = states[r * 4 + 1];
  int s2 = states[r * 4 + 2], s3 = states[r * 4 + 3];
  float chain = trans[s0 * NS_ + s1] + trans[s1 * NS_ + s2] + trans[s2 * NS_ + s3];
  int t = (r >> 2) & 7;
  zArr[r]   = (t == 0) ? (initlp[s0] + chain) : chain;
  sfA[r]    = s0;  slA[r] = s3;
  stall[r][0] = s0; stall[r][1] = s1; stall[r][2] = s2; stall[r][3] = s3;
  obsArr[r] = fwd[r];
  if (r < 16) beta[r] = 0.0f;
  __syncthreads();

  for (int u = 6; u >= 0; --u) {                    // reverse scan over T-1 steps
    if (r < 16) {
      int b = r >> 2, ki = r & 3;
      int ri = (b * 8 + u) * 4 + ki;
      float vals[4]; bool ok[4];
      float m = NEGINF;
#pragma unroll
      for (int j = 0; j < 4; ++j) {
        int rj = (b * 8 + u + 1) * 4 + j;
        bool ov = false;
#pragma unroll
        for (int a = 0; a < 4; ++a)
#pragma unroll
          for (int c = 0; c < 4; ++c)
            ov = ov || (stall[ri][a] == stall[rj][c]);
        ok[j] = !ov;
        if (!ov) {
          float tr = trans[slA[ri] * NS_ + sfA[rj]] + ext[slA[ri] * NS_ + sfA[rj]];
          vals[j] = beta[b * 4 + j] + obsArr[rj] + zArr[rj] + tr;
          m = fmaxf(m, vals[j]);
        }
      }
      float lse;
      if (m == NEGINF) lse = NEGINF;
      else {
        float ss = 0.0f;
#pragma unroll
        for (int j = 0; j < 4; ++j) if (ok[j]) ss += __expf(vals[j] - m);
        lse = m + __logf(ss);
      }
      nbeta[r] = lse + zArr[ri];
    }
    __syncthreads();
    if (r < 16) beta[r] = nbeta[r];
    __syncthreads();
  }

  if (r < 16) {
    int b = r >> 2, k = r & 3;
    int ri = (b * 8 + 0) * 4 + k;
    sc0[r] = beta[r] + obsArr[ri] + zArr[ri] + ext[sidp[0] * NS_ + sfA[ri]];
  }
  __syncthreads();
  if (r == 0) {
    float tot = 0.0f;
    for (int b = 0; b < 4; ++b) {
      float m = NEGINF;
      for (int k = 0; k < 4; ++k) m = fmaxf(m, sc0[b * 4 + k]);
      float ss = 0.0f;
      for (int k = 0; k < 4; ++k) ss += __expf(sc0[b * 4 + k] - m);
      tot += m + __logf(ss);
    }
    out[0] = -tot;
  }
}

// ---------------------------------------------------------------------------
extern "C" void kernel_launch(void* const* d_in, const int* in_sizes, int n_in,
                              void* d_out, int out_size, void* d_ws, size_t ws_size,
                              hipStream_t stream) {
  const float* X      = (const float*)d_in[0];   // output [4096,512] (flattened)
  const float* W      = (const float*)d_in[1];   // [512, 32000]
  const float* bias   = (const float*)d_in[2];   // [32000]
  const int*   target = (const int*)d_in[3];     // [4096]
  const int*   tgtidx = (const int*)d_in[4];     // [128,2]
  const int*   states = (const int*)d_in[5];     // [4,8,4,4]
  const float* initlp = (const float*)d_in[6];   // [128]
  const float* trans  = (const float*)d_in[7];   // [128,128]
  const float* ext    = (const float*)d_in[8];   // [128,128]
  const int*   sid    = (const int*)d_in[9];     // scalar
  float* out = (float*)d_out;

  char* wsb = (char*)d_ws;
  __bf16* Wfrag = (__bf16*)wsb;                              // 32,768,000 B
  float*  partM = (float*)(wsb + (size_t)NCT_ * KSTEPS_ * 32 * 16 * 2);
  float*  partS = partM + (size_t)ROWS_ * PARTS_;
  float*  psk   = partS + (size_t)ROWS_ * PARTS_;
  float*  fwd   = psk + ROWS_;

  hmm_pack_w<<<(NCT_ * KSTEPS_ * 32 + 255) / 256, 256, 0, stream>>>(W, Wfrag);
  hmm_gemm_softmax<<<dim3(ROWS_ / MB_, NSPLIT_), 256, 0, stream>>>(X, bias, Wfrag, partM, partS);
  hmm_merge_target<<<ROWS_ / 256, 256, 0, stream>>>(X, W, bias, target, partM, partS, psk);
  hmm_span_sum<<<1, 128, 0, stream>>>(psk, tgtidx, fwd);
  hmm_final<<<1, 128, 0, stream>>>(fwd, states, initlp, trans, ext, sid, out);
}
